// MyGRUModel_82274393522679
// MI455X (gfx1250) — matrix operand order repaired and checked by reading, not compiled, not run on hardware
//
#include <hip/hip_runtime.h>
#include <hip/hip_bf16.h>

typedef __attribute__((ext_vector_type(16))) _Float16 v16h;
typedef __attribute__((ext_vector_type(8)))  _Float16 v8h;
typedef __attribute__((ext_vector_type(8)))  float    v8f;

// ---------------------------------------------------------------------------
// Elementwise f32 -> f16 convert (grid-stride)
// ---------------------------------------------------------------------------
__global__ void f32_to_f16_kernel(const float* __restrict__ in,
                                  _Float16* __restrict__ out, size_t n) {
  size_t i = (size_t)blockIdx.x * blockDim.x + threadIdx.x;
  size_t stride = (size_t)gridDim.x * blockDim.x;
  for (; i < n; i += stride) out[i] = (_Float16)in[i];
}

// ---------------------------------------------------------------------------
// Embedding gather: x16[row, c] = (f16) encoder_w[ids[row], c]
// ---------------------------------------------------------------------------
__global__ void embed_kernel(const int* __restrict__ ids,
                             const float* __restrict__ enc,
                             _Float16* __restrict__ x16,
                             int rows, int E) {
  int idx = blockIdx.x * blockDim.x + threadIdx.x;
  if (idx >= rows * E) return;
  int row = idx / E;
  int c   = idx - row * E;
  x16[idx] = (_Float16)enc[(size_t)ids[row] * E + c];
}

// ---------------------------------------------------------------------------
// WMMA GEMM: C[M,N] = A[M,K] * W[N,K]^T (+ bias[N])
// A, W row-major f16; C row-major f32 (ldc == N).
// One wave -> 16x64 output strip (4 N-tiles, A regs reused 4x).
// blockDim = 256 (8 waves). grid.x covers N/64 groups / 8, grid.y = M/16.
// ---------------------------------------------------------------------------
__global__ void gemm_wmma_f16(const _Float16* __restrict__ A,
                              const _Float16* __restrict__ W,
                              const float* __restrict__ bias,
                              float* __restrict__ C,
                              int M, int N, int K) {
  const int lane  = threadIdx.x & 31;
  const int wave  = threadIdx.x >> 5;
  const int mBase = blockIdx.y << 4;
  const int nBase = (blockIdx.x * 8 + wave) << 6;   // 64-wide strip per wave
  if (mBase >= M || nBase >= N) return;             // wave-uniform: EXEC stays all-1s

  const int half = lane >> 4;                        // lane-half per WMMA layout
  const int lid  = lane & 15;                        // A row-in-tile / B,C col-in-tile

  v8f acc0 = {}, acc1 = {}, acc2 = {}, acc3 = {};
  const _Float16* __restrict__ aRow = A + (size_t)(mBase + lid) * K;
  const _Float16* __restrict__ w0 = W + (size_t)(nBase      + lid) * K;
  const _Float16* __restrict__ w1 = W + (size_t)(nBase + 16 + lid) * K;
  const _Float16* __restrict__ w2 = W + (size_t)(nBase + 32 + lid) * K;
  const _Float16* __restrict__ w3 = W + (size_t)(nBase + 48 + lid) * K;

  for (int k0 = 0; k0 < K; k0 += 32) {
    // A 16x32 f16 layout: elem j -> K = k0 + 16*half + j (contiguous, matches B)
    v8h a0 = *(const v8h*)(aRow + k0 + half * 16);
    v8h a1 = *(const v8h*)(aRow + k0 + 8 + half * 16);
    v16h a;
#pragma unroll
    for (int i = 0; i < 8; ++i) { a[i] = a0[i]; a[8 + i] = a1[i]; }
    if (k0 + 32 < K) __builtin_prefetch(aRow + k0 + 64, 0, 1);

    // B 32x16 f16 layout: elem j -> K = k0 + 16*half + j (contiguous 32B per lane)
    const int kb = k0 + half * 16;
    v16h b0 = *(const v16h*)(w0 + kb);
    v16h b1 = *(const v16h*)(w1 + kb);
    v16h b2 = *(const v16h*)(w2 + kb);
    v16h b3 = *(const v16h*)(w3 + kb);

    acc0 = __builtin_amdgcn_wmma_f32_16x16x32_f16(false, a, false, b0, (short)0, acc0, false, false);
    acc1 = __builtin_amdgcn_wmma_f32_16x16x32_f16(false, a, false, b1, (short)0, acc1, false, false);
    acc2 = __builtin_amdgcn_wmma_f32_16x16x32_f16(false, a, false, b2, (short)0, acc2, false, false);
    acc3 = __builtin_amdgcn_wmma_f32_16x16x32_f16(false, a, false, b3, (short)0, acc3, false, false);
  }

  // C/D layout: element v of v8f -> row m = mBase + v + 8*half, col n = nBase + t*16 + lid
  v8f accs[4] = {acc0, acc1, acc2, acc3};
#pragma unroll
  for (int t = 0; t < 4; ++t) {
    const int n = nBase + t * 16 + lid;
    const float bv = bias ? bias[n] : 0.0f;
#pragma unroll
    for (int v = 0; v < 8; ++v) {
      const int m = mBase + v + half * 8;
      C[(size_t)m * N + n] = accs[t][v] + bv;
    }
  }
}

// ---------------------------------------------------------------------------
// Device-scope grid barrier (monotonic counter + generation flag, both zeroed
// by hipMemsetAsync before launch -> deterministic across graph replays).
// ---------------------------------------------------------------------------
__device__ inline void grid_barrier(unsigned* counter, unsigned* gen,
                                    unsigned nblk, unsigned phase) {
  __syncthreads();
  if (threadIdx.x == 0) {
    __threadfence();  // make this block's stores device-visible
    unsigned prev = __hip_atomic_fetch_add(counter, 1u, __ATOMIC_ACQ_REL,
                                           __HIP_MEMORY_SCOPE_AGENT);
    if (prev == phase * nblk + (nblk - 1)) {
      __hip_atomic_store(gen, phase + 1u, __ATOMIC_RELEASE,
                         __HIP_MEMORY_SCOPE_AGENT);
    } else {
      while (__hip_atomic_load(gen, __ATOMIC_ACQUIRE,
                               __HIP_MEMORY_SCOPE_AGENT) <= phase) {
        __builtin_amdgcn_s_sleep(1);
      }
    }
  }
  __syncthreads();
}

// ---------------------------------------------------------------------------
// Persistent fused GRU recurrence for one layer (single launch, all S steps).
// Grid: 48 blocks x 256 threads = 384 waves == 2 M-tiles x 192 N-tiles of gh.
// Per step: phase A: gh = h16 @ Wl^T via WMMA (one 16x16 tile/wave);
//           barrier; phase B: gate math updates hbuf/h16 and writes y16;
//           barrier (protects gh + h16 reuse next step).
// Each wave's 16 W rows (32KB) stay hot in L0 across all 128 steps.
// ---------------------------------------------------------------------------
__global__ void gru_recurrence_fused(const _Float16* __restrict__ Wl,   // (3H,H) f16
                                     const float* __restrict__ gi_all,  // (S,B,3H)
                                     float* __restrict__ hbuf,          // (B,H) master
                                     _Float16* __restrict__ h16,        // (B,H)
                                     _Float16* __restrict__ y16,        // (S,B,H)
                                     float* __restrict__ gh,            // (B,3H)
                                     unsigned* __restrict__ bstate,     // [counter,gen]
                                     int steps) {
  constexpr int H = 1024, G = 3 * H, Bz = 32;
  const int lane = threadIdx.x & 31;
  const int wave = threadIdx.x >> 5;
  const int half = lane >> 4;
  const int lid  = lane & 15;
  const int wgid  = blockIdx.x * 8 + wave;  // 0..383 (exact fit, no guards)
  const int nTile = wgid >> 1;              // 0..191
  const int mTile = wgid & 1;               // 0..1
  const _Float16* __restrict__ wr = Wl  + (size_t)(nTile * 16 + lid) * H;
  const _Float16* __restrict__ ar = h16 + (size_t)(mTile * 16 + lid) * H;
  unsigned* counter = bstate;
  unsigned* gen     = bstate + 1;
  const unsigned nblk = gridDim.x;
  unsigned phase = 0;

  for (int s = 0; s < steps; ++s) {
    // ---- phase A: one 16x16 tile of gh per wave ----
    v8f acc = {};
    for (int k0 = 0; k0 < H; k0 += 32) {
      v8h a0 = *(const v8h*)(ar + k0 + half * 16);
      v8h a1 = *(const v8h*)(ar + k0 + 8 + half * 16);
      v16h a;
#pragma unroll
      for (int i = 0; i < 8; ++i) { a[i] = a0[i]; a[8 + i] = a1[i]; }
      v16h b = *(const v16h*)(wr + k0 + half * 16);
      acc = __builtin_amdgcn_wmma_f32_16x16x32_f16(false, a, false, b,
                                                   (short)0, acc, false, false);
    }
    {
      const int n = nTile * 16 + lid;
#pragma unroll
      for (int v = 0; v < 8; ++v) {
        const int m = mTile * 16 + v + half * 8;
        gh[(size_t)m * G + n] = acc[v];
      }
    }
    grid_barrier(counter, gen, nblk, phase); ++phase;

    // ---- phase B: gate fusion over B*H elements (grid-stride) ----
    const float* gis = gi_all + (size_t)s * Bz * G;
    _Float16* ys = y16 + (size_t)s * Bz * H;
    for (int idx = blockIdx.x * blockDim.x + threadIdx.x; idx < Bz * H;
         idx += (int)(nblk * blockDim.x)) {
      int b = idx >> 10;           // / H
      int j = idx & (H - 1);       // % H
      const float* gib = gis + (size_t)b * G;
      const float* ghb = gh  + (size_t)b * G;
      float ir = gib[j], ii = gib[H + j], in_ = gib[2 * H + j];
      float hr = ghb[j], hi = ghb[H + j], hn  = ghb[2 * H + j];
      float r = 1.0f / (1.0f + __expf(-(ir + hr)));
      float z = 1.0f / (1.0f + __expf(-(ii + hi)));
      float nn = tanhf(in_ + r * hn);
      float hnew = nn + z * (hbuf[idx] - nn);
      hbuf[idx] = hnew;
      _Float16 hh = (_Float16)hnew;
      h16[idx] = hh;
      ys[idx]  = hh;
    }
    grid_barrier(counter, gen, nblk, phase); ++phase;
  }
}

// ---------------------------------------------------------------------------
// Host-side orchestration
// ---------------------------------------------------------------------------
extern "C" void kernel_launch(void* const* d_in, const int* in_sizes, int n_in,
                              void* d_out, int out_size, void* d_ws, size_t ws_size,
                              hipStream_t stream) {
  (void)in_sizes; (void)n_in; (void)out_size; (void)ws_size;
  constexpr int V = 32000, E = 1024, H = 1024, L = 2, S = 128, B = 32;
  constexpr int M = S * B;          // 4096 rows
  constexpr int G = 3 * H;          // 3072 gate width
  constexpr int RGRID = 48;         // persistent-recurrence blocks (trivially co-resident)

  const int*   ids    = (const int*)d_in[0];
  const float* hidden = (const float*)d_in[1];   // (L,B,H)
  const float* enc    = (const float*)d_in[2];   // (V,E)
  const float* w_ih   = (const float*)d_in[3];   // (L,3H,H)
  const float* w_hh   = (const float*)d_in[4];   // (L,3H,H)
  const float* b_ih   = (const float*)d_in[5];   // (L,3H)
  const float* dec_w  = (const float*)d_in[6];   // (V,H)
  const float* dec_b  = (const float*)d_in[7];   // (V,)
  float* out = (float*)d_out;                    // logits (S,B,V) ++ hidden (L,B,H)

  // ---- workspace carve-out (256B-aligned chunks) ----
  char* p = (char*)d_ws;
  auto carve = [&](size_t bytes) -> void* {
    void* r = (void*)p;
    p += (bytes + 255) & ~(size_t)255;
    return r;
  };
  _Float16* w_ih16  = (_Float16*)carve((size_t)L * G * H * 2);
  _Float16* w_hh16  = (_Float16*)carve((size_t)L * G * H * 2);
  _Float16* dec_w16 = (_Float16*)carve((size_t)V * H * 2);
  _Float16* x16[3];
  x16[0] = (_Float16*)carve((size_t)M * H * 2);
  x16[1] = (_Float16*)carve((size_t)M * H * 2);
  x16[2] = (_Float16*)carve((size_t)M * H * 2);
  float*    gi_all = (float*)carve((size_t)M * G * 4);
  float*    gh     = (float*)carve((size_t)B * G * 4);
  float*    hbuf   = (float*)carve((size_t)B * H * 4);
  _Float16* h16    = (_Float16*)carve((size_t)B * H * 2);
  unsigned* bstate = (unsigned*)carve(sizeof(unsigned) * 2 * L);  // per-layer barrier state

  const dim3 blk(256);

  // zero barrier state once per launch (graph-capturable memset node)
  hipMemsetAsync(bstate, 0, sizeof(unsigned) * 2 * L, stream);

  // ---- weight conversion (once per launch, deterministic) ----
  f32_to_f16_kernel<<<dim3(2048), blk, 0, stream>>>(w_ih,  w_ih16,  (size_t)L * G * H);
  f32_to_f16_kernel<<<dim3(2048), blk, 0, stream>>>(w_hh,  w_hh16,  (size_t)L * G * H);
  f32_to_f16_kernel<<<dim3(4096), blk, 0, stream>>>(dec_w, dec_w16, (size_t)V * H);

  // ---- embedding gather ----
  embed_kernel<<<dim3((M * E + 255) / 256), blk, 0, stream>>>(ids, enc, x16[0], M, E);

  // ---- layers ----
  for (int l = 0; l < L; ++l) {
    const _Float16* xin  = x16[l];
    _Float16*       xout = x16[l + 1];

    // gi_all = x @ w_ih[l]^T + b_ih[l]   (4096 x 3072, K=1024)
    gemm_wmma_f16<<<dim3(G / 64 / 8, M / 16), blk, 0, stream>>>(
        xin, w_ih16 + (size_t)l * G * H, b_ih + (size_t)l * G, gi_all, M, G, H);

    // h = hidden[l]
    hipMemcpyAsync(hbuf, hidden + (size_t)l * B * H, (size_t)B * H * 4,
                   hipMemcpyDeviceToDevice, stream);
    f32_to_f16_kernel<<<dim3(128), blk, 0, stream>>>(
        hidden + (size_t)l * B * H, h16, (size_t)B * H);

    // all 128 recurrence steps in one persistent launch
    gru_recurrence_fused<<<dim3(RGRID), blk, 0, stream>>>(
        w_hh16 + (size_t)l * G * H, gi_all, hbuf, h16, xout, gh,
        bstate + 2 * l, S);

    // next_hidden[l] = h
    hipMemcpyAsync(out + (size_t)S * B * V + (size_t)l * B * H, hbuf,
                   (size_t)B * H * 4, hipMemcpyDeviceToDevice, stream);
  }

  // ---- decoder: logits = x @ dec_w^T + dec_b  (4096 x 32000, K=1024) ----
  gemm_wmma_f16<<<dim3((V / 64 + 7) / 8, M / 16), blk, 0, stream>>>(
      x16[2], dec_w16, dec_b, out, M, V, H);
}
